// TriangleAttention_55688545960632
// MI455X (gfx1250) — matrix operand order repaired and checked by reading, not compiled, not run on hardware
//
#include <hip/hip_runtime.h>

typedef __attribute__((ext_vector_type(16))) _Float16 v16h;
typedef __attribute__((ext_vector_type(8)))  _Float16 v8h;
typedef __attribute__((ext_vector_type(8)))  float    v8f;

#define S   384
#define CZ  128
#define H   4
#define CH  32
#define NROW (S*S)      // 147456 flattened (i,j) rows
#define JT  (S/16)      // 24 tiles of 16 along j or k
#define RT  (NROW/16)   // 9216 row tiles

// Build a 16x32 f16 A-operand from two contiguous 8-half chunks.
// A layout as built here: lane<16 holds row (lane&15), K=0..15;
// lane>=16 holds same row set, K=16..31 (contiguous; B uses the same order).
static __device__ __forceinline__ v16h a_combine(v8h lo, v8h hi) {
  v16h a;
#pragma unroll
  for (int x = 0; x < 8; ++x) { a[x] = lo[x]; a[x + 8] = hi[x]; }
  return a;
}

static __device__ __forceinline__ v8f wmma16x16x32(v16h a, v16h b, v8f c) {
  return __builtin_amdgcn_wmma_f32_16x16x32_f16(false, a, false, b, (short)0, c,
                                                false, false);
}

// ---------------------------------------------------------------------------
// K0: convert weights to f16. W_all rows: [0,128)=Wq*scale, [128,160)=Wk,
// [160,192)=Wv, [192,320)=Wg. Also Wo -> f16. (Wb skipped: softmax-invariant.)
// ---------------------------------------------------------------------------
__global__ __launch_bounds__(256) void k0_wcvt(
    const float* __restrict__ Wq, const float* __restrict__ Wk,
    const float* __restrict__ Wv, const float* __restrict__ Wg,
    const float* __restrict__ Wo, _Float16* __restrict__ W_all,
    _Float16* __restrict__ Wo_h) {
  int idx = blockIdx.x * 256 + threadIdx.x;
  const float scale = 0.1767766952966369f;  // 1/sqrt(CH)
  if (idx < 320 * CZ) {
    int r = idx / CZ, c = idx % CZ;
    float v;
    if (r < 128)      v = Wq[r * CZ + c] * scale;
    else if (r < 160) v = Wk[(r - 128) * CZ + c];
    else if (r < 192) v = Wv[(r - 160) * CZ + c];
    else              v = Wg[(r - 192) * CZ + c];
    W_all[idx] = (_Float16)v;
  } else if (idx < 320 * CZ + CZ * CZ) {
    int j = idx - 320 * CZ;
    Wo_h[j] = (_Float16)Wo[j];
  }
}

// ---------------------------------------------------------------------------
// K1: LayerNorm + fused projections via WMMA. One wave per 16 rows.
// Outputs: q[row][128] f16 (scale folded in), kf[row][32] f16,
// vt[i][c][j] f16 (transposed V), g[row][128] f16 (sigmoid applied).
// ---------------------------------------------------------------------------
__global__ __launch_bounds__(32) void k1_ln_proj(
    const float* __restrict__ z, const float* __restrict__ ln_w,
    const float* __restrict__ ln_b, const _Float16* __restrict__ W_all,
    const float* __restrict__ bg, _Float16* __restrict__ q,
    _Float16* __restrict__ kf, _Float16* __restrict__ vt,
    _Float16* __restrict__ g) {
  __shared__ __align__(16) _Float16 znh[16][CZ];
  int lane = threadIdx.x;
  int lo16 = lane & 15, hi = lane >> 4;
  int r0 = blockIdx.x * 16;

  float4 w4 = ((const float4*)ln_w)[lane];
  float4 b4 = ((const float4*)ln_b)[lane];
  for (int r = 0; r < 16; ++r) {
    int row = r0 + r;
    float4 zv = ((const float4*)(z + (size_t)row * CZ))[lane];
    float s  = zv.x + zv.y + zv.z + zv.w;
    float s2 = zv.x * zv.x + zv.y * zv.y + zv.z * zv.z + zv.w * zv.w;
#pragma unroll
    for (int off = 16; off >= 1; off >>= 1) {
      s  += __shfl_xor(s,  off, 32);
      s2 += __shfl_xor(s2, off, 32);
    }
    float mu  = s * (1.0f / CZ);
    float inv = rsqrtf(s2 * (1.0f / CZ) - mu * mu + 1e-5f);
    _Float16* dst = &znh[r][lane * 4];
    dst[0] = (_Float16)((zv.x - mu) * inv * w4.x + b4.x);
    dst[1] = (_Float16)((zv.y - mu) * inv * w4.y + b4.y);
    dst[2] = (_Float16)((zv.z - mu) * inv * w4.z + b4.z);
    dst[3] = (_Float16)((zv.w - mu) * inv * w4.w + b4.w);
  }
  __syncthreads();

  // Preload the 4 A tiles (zn 16 rows x 32 K-chunk).
  v16h a[4];
#pragma unroll
  for (int t = 0; t < 4; ++t)
    a[t] = a_combine(*(const v8h*)&znh[lo16][t * 32 + hi * 16],
                     *(const v8h*)&znh[lo16][t * 32 + hi * 16 + 8]);

  for (int n = 0; n < 20; ++n) {
    int n0 = n * 16;
    v8f acc = {};
#pragma unroll
    for (int t = 0; t < 4; ++t) {
      v16h b = *(const v16h*)&W_all[(size_t)(n0 + lo16) * CZ + t * 32 + hi * 16];
      acc = wmma16x16x32(a[t], b, acc);
    }
    int out = n0 + lo16;
#pragma unroll
    for (int vv = 0; vv < 8; ++vv) {
      int row = r0 + vv + hi * 8;
      float val = acc[vv];
      if (n0 < 128) {
        q[(size_t)row * 128 + out] = (_Float16)val;
      } else if (n0 < 160) {
        kf[(size_t)row * 32 + (out - 128)] = (_Float16)val;
      } else if (n0 < 192) {
        int i = row / S, j = row % S;
        vt[((size_t)i * CH + (out - 160)) * S + j] = (_Float16)val;
      } else {
        float gv = val + bg[out - 192];
        gv = 1.0f / (1.0f + __expf(-gv));
        g[(size_t)row * 128 + (out - 192)] = (_Float16)gv;
      }
    }
  }
}

// ---------------------------------------------------------------------------
// K2: attention. One wave per (j-tile, head, i). logits -> softmax -> @V,
// gate, store gated output f16. Bias term omitted (softmax-invariant).
// ---------------------------------------------------------------------------
__global__ __launch_bounds__(32) void k2_attn(
    const _Float16* __restrict__ q, const _Float16* __restrict__ kf,
    const _Float16* __restrict__ vt, const _Float16* __restrict__ g,
    _Float16* __restrict__ go) {
  __shared__ __align__(16) float    lg[16][S + 4];    // pad: stride 388 dwords
  __shared__ __align__(16) _Float16 wh[16][S + 16];   // pad: stride 400 halves
  int lane = threadIdx.x;
  int lo16 = lane & 15, hi = lane >> 4;
  int jt = blockIdx.x, h = blockIdx.y, i = blockIdx.z;

  // A tile: q rows jt*16.., head h (scale pre-folded).
  const _Float16* qrow = q + ((size_t)(i * S) + jt * 16 + lo16) * 128 + h * CH;
  v16h aq = a_combine(*(const v8h*)(qrow + hi * 16),
                      *(const v8h*)(qrow + hi * 16 + 8));

  // logits: 24 B-tiles of K^T (B[c][kk] = K[kk][c], contiguous per lane).
  for (int kt = 0; kt < JT; ++kt) {
    const _Float16* krow = kf + ((size_t)(i * S) + kt * 16 + lo16) * CH + hi * 16;
    v16h bk = *(const v16h*)krow;
    v8f d = {};
    d = wmma16x16x32(aq, bk, d);
#pragma unroll
    for (int vv = 0; vv < 8; ++vv)
      lg[vv + hi * 8][kt * 16 + lo16] = d[vv];
  }
  __syncthreads();

  // softmax over 384 keys; 2 lanes per row (lane, lane+16).
  float mx = -3.0e38f;
  for (int t = 0; t < 192; ++t) mx = fmaxf(mx, lg[lo16][hi * 192 + t]);
  mx = fmaxf(mx, __shfl_xor(mx, 16, 32));
  float sum = 0.f;
  for (int t = 0; t < 192; ++t) {
    float e = __expf(lg[lo16][hi * 192 + t] - mx);
    lg[lo16][hi * 192 + t] = e;
    sum += e;
  }
  sum += __shfl_xor(sum, 16, 32);
  float inv = 1.0f / sum;
  for (int t = 0; t < 192; ++t)
    wh[lo16][hi * 192 + t] = (_Float16)(lg[lo16][hi * 192 + t] * inv);
  __syncthreads();

  // o = W @ V : K-dim 384 in 12 chunks of 32; N = 32 -> two accumulators.
  v8f acc0 = {}, acc1 = {};
  for (int t = 0; t < 12; ++t) {
    v16h aw = a_combine(*(const v8h*)&wh[lo16][t * 32 + hi * 16],
                        *(const v8h*)&wh[lo16][t * 32 + hi * 16 + 8]);
    const _Float16* v0 = vt + ((size_t)i * CH + lo16) * S + t * 32 + hi * 16;
    const _Float16* v1 = vt + ((size_t)i * CH + 16 + lo16) * S + t * 32 + hi * 16;
    v16h b0 = *(const v16h*)v0;
    v16h b1 = *(const v16h*)v1;
    acc0 = wmma16x16x32(aw, b0, acc0);
    acc1 = wmma16x16x32(aw, b1, acc1);
  }

  // gate and store
#pragma unroll
  for (int vv = 0; vv < 8; ++vv) {
    int j = jt * 16 + vv + hi * 8;
    size_t base = ((size_t)(i * S) + j) * 128 + h * CH;
    float g0 = (float)g[base + lo16];
    float g1 = (float)g[base + 16 + lo16];
    go[base + lo16]      = (_Float16)(acc0[vv] * g0);
    go[base + 16 + lo16] = (_Float16)(acc1[vv] * g1);
  }
}

// ---------------------------------------------------------------------------
// K3: output projection: out = go @ Wo^T + bo  (f32 output).
// ---------------------------------------------------------------------------
__global__ __launch_bounds__(32) void k3_outproj(
    const _Float16* __restrict__ go, const _Float16* __restrict__ Wo_h,
    const float* __restrict__ bo, float* __restrict__ out) {
  int lane = threadIdx.x;
  int lo16 = lane & 15, hi = lane >> 4;
  int r0 = blockIdx.x * 16;
  const _Float16* grow = go + (size_t)(r0 + lo16) * 128;
  v16h a[4];
#pragma unroll
  for (int t = 0; t < 4; ++t)
    a[t] = a_combine(*(const v8h*)(grow + t * 32 + hi * 16),
                     *(const v8h*)(grow + t * 32 + hi * 16 + 8));
  for (int n = 0; n < 8; ++n) {
    v8f acc = {};
#pragma unroll
    for (int t = 0; t < 4; ++t) {
      v16h b = *(const v16h*)&Wo_h[(size_t)(n * 16 + lo16) * CZ + t * 32 + hi * 16];
      acc = wmma16x16x32(a[t], b, acc);
    }
    int co = n * 16 + lo16;
    float bias = bo[co];
#pragma unroll
    for (int vv = 0; vv < 8; ++vv) {
      int row = r0 + vv + hi * 8;
      out[(size_t)row * CZ + co] = acc[vv] + bias;
    }
  }
}

// ---------------------------------------------------------------------------
extern "C" void kernel_launch(void* const* d_in, const int* in_sizes, int n_in,
                              void* d_out, int out_size, void* d_ws,
                              size_t ws_size, hipStream_t stream) {
  (void)in_sizes; (void)n_in; (void)out_size; (void)ws_size;
  const float* z    = (const float*)d_in[0];
  const float* ln_w = (const float*)d_in[1];
  const float* ln_b = (const float*)d_in[2];
  const float* Wq   = (const float*)d_in[3];
  const float* Wk   = (const float*)d_in[4];
  const float* Wv   = (const float*)d_in[5];
  // d_in[6] = Wb: its bias broadcasts along the key/softmax axis in the
  // reference ("softmax-invariant") -> zero effect on output; skipped.
  const float* Wg   = (const float*)d_in[7];
  const float* bg   = (const float*)d_in[8];
  const float* Wo   = (const float*)d_in[9];
  const float* bo   = (const float*)d_in[10];
  float* out = (float*)d_out;

  char* ws = (char*)d_ws;
  _Float16* W_all = (_Float16*)ws;  ws += (size_t)320 * CZ * 2;      //  80 KB
  _Float16* Wo_h  = (_Float16*)ws;  ws += (size_t)CZ * CZ * 2;       //  32 KB
  _Float16* q     = (_Float16*)ws;  ws += (size_t)NROW * 128 * 2;    // 37.7 MB
  _Float16* kf    = (_Float16*)ws;  ws += (size_t)NROW * CH * 2;     //  9.4 MB
  _Float16* vt    = (_Float16*)ws;  ws += (size_t)S * CH * S * 2;    //  9.4 MB
  _Float16* g     = (_Float16*)ws;  ws += (size_t)NROW * 128 * 2;    // 37.7 MB
  _Float16* go    = (_Float16*)ws;                                   // 37.7 MB

  k0_wcvt<<<(320 * CZ + CZ * CZ + 255) / 256, 256, 0, stream>>>(
      Wq, Wk, Wv, Wg, Wo, W_all, Wo_h);
  k1_ln_proj<<<RT, 32, 0, stream>>>(z, ln_w, ln_b, W_all, bg, q, kf, vt, g);
  dim3 g2(JT, H, S);
  k2_attn<<<g2, 32, 0, stream>>>(q, kf, vt, g, go);
  k3_outproj<<<RT, 32, 0, stream>>>(go, Wo_h, bo, out);
}
